// MAMoE_40364102648320
// MI455X (gfx1250) — matrix operand order repaired and checked by reading, not compiled, not run on hardware
//
#include <hip/hip_runtime.h>
#include <hip/hip_bf16.h>

typedef __attribute__((ext_vector_type(16))) _Float16 v16h;
typedef __attribute__((ext_vector_type(8)))  _Float16 v8h;
typedef __attribute__((ext_vector_type(8)))  float    v8f;

#define NTOK  131072   // 8*128*128
#define CDIM  192
#define HD    64
#define HIDDEN 256
#define NEXP  5

// ---- CDNA5 async global->LDS path (guarded; falls back to wide copies) ----
#if defined(__has_builtin)
#if __has_builtin(__builtin_amdgcn_global_load_async_to_lds_b128)
#define USE_ASYNC_LDS 1
#endif
#endif
#ifndef USE_ASYNC_LDS
#define USE_ASYNC_LDS 0
#endif

#if USE_ASYNC_LDS
// builtin expects pointers to gcc-vector int4 in addrspace(1)/(3)
typedef int v4i_raw __attribute__((vector_size(4 * sizeof(int))));
typedef __attribute__((address_space(1))) v4i_raw as1_v4i;
typedef __attribute__((address_space(3))) v4i_raw as3_v4i;
static __device__ __forceinline__ as1_v4i* to_glb(const void* p) {
  return (as1_v4i*)(unsigned long long)p;
}
static __device__ __forceinline__ as3_v4i* to_lds(void* p) {
  // LDS offset lives in the low 32 bits of the flat address (ISA aperture rule)
  return (as3_v4i*)(unsigned int)(unsigned long long)p;
}
#if __has_builtin(__builtin_amdgcn_s_wait_asynccnt)
static __device__ __forceinline__ void wait_async0() {
  __builtin_amdgcn_s_wait_asynccnt(0);
}
#else
static __device__ __forceinline__ void wait_async0() {
  asm volatile("s_wait_asynccnt 0x0" ::: "memory");
}
#endif
#endif

// ---------------------------------------------------------------------------
// Wide WMMA fragment helpers (CDNA5 16x16x32 f16, wave32).
//  A-frag of lane(ln,hf): halves [k0+hf*8, +8) and [k0+16+hf*8, +8)  -> 2xB128
//  B-frag of lane(ln,hf): halves [k0+hf*8, +8) and [k0+16+hf*8, +8)  -> 2xB128
//  C/D:   vgpr r -> M = r + 8*hf ; lane&15 = N
// ---------------------------------------------------------------------------
static __device__ __forceinline__ v16h frag_a(const _Float16* p, int hf) {
  v8h lo = *(const v8h*)(p + hf * 8);
  v8h hi = *(const v8h*)(p + 16 + hf * 8);
  return __builtin_shufflevector(lo, hi, 0, 1, 2, 3, 4, 5, 6, 7, 8, 9, 10, 11,
                                 12, 13, 14, 15);
}
static __device__ __forceinline__ v16h frag_b(const _Float16* p, int hf) {
  v8h lo = *(const v8h*)(p + hf * 8);
  v8h hi = *(const v8h*)(p + 16 + hf * 8);
  return __builtin_shufflevector(lo, hi, 0, 1, 2, 3, 4, 5, 6, 7, 8, 9, 10, 11,
                                 12, 13, 14, 15);
}
static __device__ __forceinline__ v8f wmma(v16h a, v16h b, v8f c) {
  return __builtin_amdgcn_wmma_f32_16x16x32_f16(false, a, false, b, (short)0, c,
                                                false, false);
}
static __device__ __forceinline__ void load_bias8(const float* p, float* bv) {
  float4 lo = *(const float4*)p;
  float4 hi = *((const float4*)p + 1);
  bv[0] = lo.x; bv[1] = lo.y; bv[2] = lo.z; bv[3] = lo.w;
  bv[4] = hi.x; bv[5] = hi.y; bv[6] = hi.z; bv[7] = hi.w;
}

// ---------------------------------------------------------------------------
// Weight pre-swizzle: Wt[n][k] = (f16) W[k][n]   (weights become the A operand)
// ---------------------------------------------------------------------------
__global__ void wswz_kernel(const float* __restrict__ W, _Float16* __restrict__ Wt,
                            int K, int N) {
  int idx = blockIdx.x * 256 + threadIdx.x;
  if (idx >= K * N) return;
  int n = idx / K, k = idx % K;
  Wt[idx] = (_Float16)W[k * N + n];
}

// conv OIHW (co,ci,3x3|1x9|9x1) -> Wt[co][tap*64+ci] (taps always 9)
__global__ void convswz_kernel(const float* __restrict__ W,
                               _Float16* __restrict__ Wt) {
  int idx = blockIdx.x * 256 + threadIdx.x;  // 64*576
  int co = idx / 576, r = idx % 576;
  int t = r >> 6, ci = r & 63;
  Wt[idx] = (_Float16)W[(co * 64 + ci) * 9 + t];
}

__global__ void cvt16_kernel(const float* __restrict__ in,
                             _Float16* __restrict__ out) {
  long i = ((long)blockIdx.x * 256 + threadIdx.x) * 8;
#pragma unroll
  for (int e = 0; e < 8; ++e) out[i + e] = (_Float16)in[i + e];
}

__global__ void zero_counts_kernel(int* counts, int n) {
  int i = threadIdx.x;
  if (i < n) counts[i] = 0;
}

// ---------------------------------------------------------------------------
// Generic GEMM (transposed orientation, LDS-free, no barriers):
//   out[tok][col] = sum_k act[tok][k] * Wt[col][k]  (+bias[col])
// block: 64 tokens x 64 cols; 8 waves, 2 tiles each (shared A-frag).
// grid = (ntok/64, ncols/64).
// ---------------------------------------------------------------------------
__global__ __launch_bounds__(256) void gemm_tn_kernel(
    const _Float16* __restrict__ act, int lda, const _Float16* __restrict__ wt,
    const float* __restrict__ bias, void* __restrict__ outp, int ldc,
    int coloff, int K, int out_f16) {
  const int tid = threadIdx.x;
  const int lane = tid & 31;
  const int wave = tid >> 5;
  const int ln = lane & 15, hf = lane >> 4;
  const long tok0 = (long)blockIdx.x * 64;
  const int col_blk = blockIdx.y * 64;
  const int tm = wave & 3;
  const int tn0 = (wave >> 2) * 2;

  const _Float16* arow = wt + (long)(col_blk + tm * 16 + ln) * K;
  const _Float16* brow0 = act + (tok0 + tn0 * 16 + ln) * lda;
  const _Float16* brow1 = brow0 + 16 * lda;

  v8f acc0 = {}, acc1 = {};
  for (int k0 = 0; k0 < K; k0 += 32) {
    v16h a = frag_a(arow + k0, hf);
    v16h b0 = frag_b(brow0 + k0, hf);
    v16h b1 = frag_b(brow1 + k0, hf);
    acc0 = wmma(a, b0, acc0);
    acc1 = wmma(a, b1, acc1);
  }

  float bv[8] = {0, 0, 0, 0, 0, 0, 0, 0};
  if (bias) load_bias8(bias + col_blk + tm * 16 + 8 * hf, bv);
  const long tokA = tok0 + tn0 * 16 + ln;
  const long tokB = tokA + 16;
  const int colbase = coloff + col_blk + tm * 16 + 8 * hf;

  if (out_f16) {
    _Float16* O = (_Float16*)outp;
    v8h h0, h1;
#pragma unroll
    for (int r = 0; r < 8; ++r) {
      h0[r] = (_Float16)(acc0[r] + bv[r]);
      h1[r] = (_Float16)(acc1[r] + bv[r]);
    }
    *(v8h*)(O + tokA * ldc + colbase) = h0;
    *(v8h*)(O + tokB * ldc + colbase) = h1;
  } else {
    float* O = (float*)outp;
    *(float4*)(O + tokA * ldc + colbase) =
        make_float4(acc0[0] + bv[0], acc0[1] + bv[1], acc0[2] + bv[2],
                    acc0[3] + bv[3]);
    *(float4*)(O + tokA * ldc + colbase + 4) =
        make_float4(acc0[4] + bv[4], acc0[5] + bv[5], acc0[6] + bv[6],
                    acc0[7] + bv[7]);
    *(float4*)(O + tokB * ldc + colbase) =
        make_float4(acc1[0] + bv[0], acc1[1] + bv[1], acc1[2] + bv[2],
                    acc1[3] + bv[3]);
    *(float4*)(O + tokB * ldc + colbase + 4) =
        make_float4(acc1[4] + bv[4], acc1[5] + bv[5], acc1[6] + bv[6],
                    acc1[7] + bv[7]);
  }
}

// ---------------------------------------------------------------------------
// Implicit-GEMM conv (transposed orientation): weights = A (pre-swizzled
// [co][tap*64+ci]), zero-filled patch tile = B (LDS; async-staged).
// block = 64 x-positions of one (b,y); grid.x = 2048.
// ---------------------------------------------------------------------------
__global__ __launch_bounds__(256) void conv_tn_kernel(
    const _Float16* __restrict__ X, int chan0, const _Float16* __restrict__ wt,
    const float* __restrict__ bias, _Float16* __restrict__ C, int coloff,
    int kh, int kw, int dh, int dw, int ph, int pw) {
  __shared__ __align__(16) _Float16 sB[64 * 72];
  const int tid = threadIdx.x;
  const int lane = tid & 31, wave = tid >> 5;
  const int ln = lane & 15, hf = lane >> 4;
  const int tm = wave & 3;
  const int tn0 = (wave >> 2) * 2;
  const int bx = blockIdx.x;
  const int xblk = bx & 1;
  const int y = (bx >> 1) & 127;
  const int bb = bx >> 8;
  const int x0 = xblk * 64;
  const long tokbase = ((long)bb * 128 + y) * 128;
  const int taps = kh * kw;

  v8f acc0 = {}, acc1 = {};
  for (int t = 0; t < taps; ++t) {
    int th = t / kw, tw = t % kw;
    int yy = y + th * dh - ph;
    bool yok = (unsigned)yy < 128u;
    {  // stage 64x64 patch rows (zero-filled boundary, async interior)
      int r = tid >> 2, cb = (tid & 3) * 16;
      int xx = x0 + r + tw * dw - pw;
      bool ok = yok && ((unsigned)xx < 128u);
      const _Float16* g =
          X + (((long)bb * 128 + yy) * 128 + xx) * CDIM + chan0 + cb;
      _Float16* l = sB + r * 72 + cb;
#if USE_ASYNC_LDS
      if (ok) {
        __builtin_amdgcn_global_load_async_to_lds_b128(to_glb(g), to_lds(l), 0, 0);
        __builtin_amdgcn_global_load_async_to_lds_b128(to_glb(g), to_lds(l), 16, 0);
      } else {
        v8h z = {};
        *(v8h*)l = z;
        *(v8h*)(l + 8) = z;
      }
      wait_async0();
#else
      v8h lo = {}, hi = {};
      if (ok) {
        lo = *(const v8h*)g;
        hi = *(const v8h*)(g + 8);
      }
      *(v8h*)l = lo;
      *(v8h*)(l + 8) = hi;
#endif
    }
    __syncthreads();
#pragma unroll
    for (int ch = 0; ch < 2; ++ch) {
      v16h a = frag_a(wt + (tm * 16 + ln) * 576 + t * 64 + ch * 32, hf);
      v16h b0 = frag_b(sB + (tn0 * 16 + ln) * 72 + ch * 32, hf);
      v16h b1 = frag_b(sB + ((tn0 + 1) * 16 + ln) * 72 + ch * 32, hf);
      acc0 = wmma(a, b0, acc0);
      acc1 = wmma(a, b1, acc1);
    }
    __syncthreads();
  }
  float bv[8];
  load_bias8(bias + tm * 16 + 8 * hf, bv);
  long tokA = tokbase + x0 + tn0 * 16 + ln;
  long tokB = tokA + 16;
  int colbase = coloff + tm * 16 + 8 * hf;
  v8h h0, h1;
#pragma unroll
  for (int r = 0; r < 8; ++r) {
    h0[r] = (_Float16)(acc0[r] + bv[r]);
    h1[r] = (_Float16)(acc1[r] + bv[r]);
  }
  *(v8h*)(C + tokA * 128 + colbase) = h0;
  *(v8h*)(C + tokB * 128 + colbase) = h1;
}

// ---------------------------------------------------------------------------
// Router: 5 logits per token -> argmax gate + compaction lists.
// ---------------------------------------------------------------------------
__global__ __launch_bounds__(256) void router_kernel(
    const _Float16* __restrict__ T, const float* __restrict__ RW,
    const float* __restrict__ RB, float* __restrict__ gate,
    int* __restrict__ lists, int* __restrict__ counts, int ntok) {
  int tok = blockIdx.x * 256 + threadIdx.x;
  if (tok >= ntok) return;
  float lg[NEXP];
#pragma unroll
  for (int e = 0; e < NEXP; ++e) lg[e] = RB[e];
  const _Float16* tp = T + (long)tok * HD;
  for (int c = 0; c < HD; ++c) {
    float tv = (float)tp[c];
#pragma unroll
    for (int e = 0; e < NEXP; ++e) lg[e] += tv * RW[c * NEXP + e];
  }
  int best = 0;
  float mx = lg[0];
#pragma unroll
  for (int e = 1; e < NEXP; ++e)
    if (lg[e] > mx) { mx = lg[e]; best = e; }
  float s = 0.f;
#pragma unroll
  for (int e = 0; e < NEXP; ++e) s += __expf(lg[e] - mx);
  gate[tok] = 1.0f / s;
  int pos = atomicAdd(&counts[best], 1);
  lists[best * ntok + pos] = tok;
}

// ---------------------------------------------------------------------------
// MoE (transposed orientation, all-wide): block = (expert, 64-token tile).
// Ht = W1t x Tt (LDS, wide stores), Y = W2t x Ht, gate-scaled wide scatter.
// ---------------------------------------------------------------------------
__global__ __launch_bounds__(256) void moe_kernel(
    const _Float16* __restrict__ T, const int* __restrict__ lists,
    const int* __restrict__ counts, const float* __restrict__ gate,
    const _Float16* __restrict__ w1t, const float* __restrict__ b1,
    const _Float16* __restrict__ w2t, const float* __restrict__ b2,
    _Float16* __restrict__ out, int ntok, int tilesPerE) {
  __shared__ __align__(16) _Float16 sT[64 * 72];
  __shared__ __align__(16) _Float16 sH[64 * 264];
  __shared__ int sTok[64];
  __shared__ float sGate[64];

  const int e = blockIdx.x / tilesPerE;
  const int tile = blockIdx.x % tilesPerE;
  const int cnt = counts[e];
  const int t0 = tile * 64;
  if (t0 >= cnt) return;

  const int tid = threadIdx.x;
  const int lane = tid & 31, wave = tid >> 5;
  const int ln = lane & 15, hf = lane >> 4;

  if (tid < 64) {
    int tok = (t0 + tid < cnt) ? lists[(long)e * ntok + t0 + tid] : -1;
    sTok[tid] = tok;
    sGate[tid] = (tok >= 0) ? gate[tok] : 0.f;
  }
  __syncthreads();
  {
    int r = tid >> 2, cb = (tid & 3) * 16;
    int tok = sTok[r];
    const _Float16* g = T + (long)(tok < 0 ? 0 : tok) * HD + cb;
    _Float16* l = sT + r * 72 + cb;
#if USE_ASYNC_LDS
    if (tok >= 0) {
      __builtin_amdgcn_global_load_async_to_lds_b128(to_glb(g), to_lds(l), 0, 0);
      __builtin_amdgcn_global_load_async_to_lds_b128(to_glb(g), to_lds(l), 16, 0);
    } else {
      v8h z = {};
      *(v8h*)l = z;
      *(v8h*)(l + 8) = z;
    }
    wait_async0();
#else
    v8h lo = {}, hi = {};
    if (tok >= 0) {
      lo = *(const v8h*)g;
      hi = *(const v8h*)(g + 8);
    }
    *(v8h*)l = lo;
    *(v8h*)(l + 8) = hi;
#endif
  }
  __syncthreads();

  const _Float16* w1e = w1t + (long)e * HD * HIDDEN;  // [256][64]
  const float* b1e = b1 + e * HIDDEN;
  const _Float16* w2e = w2t + (long)e * HIDDEN * HD;  // [64][256]
  const float* b2e = b2 + e * HD;

  // Ht[hid][tok]: 16 m-tiles x 4 n-tiles, 8 tiles/wave, K=64
  for (int tt = wave; tt < 64; tt += 8) {
    int tm = tt >> 2, tn = tt & 3;
    const _Float16* ar = w1e + (tm * 16 + ln) * HD;
    const _Float16* br = sT + (tn * 16 + ln) * 72;
    v8f acc = {};
#pragma unroll
    for (int kc = 0; kc < 2; ++kc)
      acc = wmma(frag_a(ar + kc * 32, hf), frag_b(br + kc * 32, hf), acc);
    float bv[8];
    load_bias8(b1e + tm * 16 + 8 * hf, bv);
    v8h h;
#pragma unroll
    for (int r = 0; r < 8; ++r) h[r] = (_Float16)fmaxf(acc[r] + bv[r], 0.f);
    *(v8h*)(sH + (tn * 16 + ln) * 264 + tm * 16 + 8 * hf) = h;
  }
  __syncthreads();

  // Y[och][tok]: 4x4 tiles, 2/wave, K=256
  for (int tt = wave; tt < 16; tt += 8) {
    int tm = tt >> 2, tn = tt & 3;
    const _Float16* ar = w2e + (tm * 16 + ln) * HIDDEN;
    const _Float16* br = sH + (tn * 16 + ln) * 264;
    v8f acc = {};
#pragma unroll
    for (int kc = 0; kc < 8; ++kc)
      acc = wmma(frag_a(ar + kc * 32, hf), frag_b(br + kc * 32, hf), acc);
    int trow = tn * 16 + ln;
    int tok = sTok[trow];
    if (tok >= 0) {
      float g = sGate[trow];
      float bv[8];
      load_bias8(b2e + tm * 16 + 8 * hf, bv);
      v8h h;
#pragma unroll
      for (int r = 0; r < 8; ++r) h[r] = (_Float16)((acc[r] + bv[r]) * g);
      *(v8h*)(out + (long)tok * HD + tm * 16 + 8 * hf) = h;
    }
  }
}

// ---------------------------------------------------------------------------
// V transpose: vt[(b*128+y)*64 + c*128 + kpos] = qkv[(b,y,kpos)][128+c]  (f16)
// ---------------------------------------------------------------------------
__global__ __launch_bounds__(256) void vtrans_kernel(
    const _Float16* __restrict__ qkv, _Float16* __restrict__ vt) {
  __shared__ __align__(16) _Float16 sT[128 * 72];
  const int bb = blockIdx.x >> 7;
  const int y = blockIdx.x & 127;
  const long tokbase = ((long)bb * 128 + y) * 128;
  const int tid = threadIdx.x;
  {
    int r = tid >> 1, cb = (tid & 1) * 32;
    const _Float16* g = qkv + (tokbase + r) * CDIM + 2 * HD + cb;
    _Float16* l = sT + r * 72 + cb;
#if USE_ASYNC_LDS
    __builtin_amdgcn_global_load_async_to_lds_b128(to_glb(g), to_lds(l), 0, 0);
    __builtin_amdgcn_global_load_async_to_lds_b128(to_glb(g), to_lds(l), 16, 0);
    __builtin_amdgcn_global_load_async_to_lds_b128(to_glb(g), to_lds(l), 32, 0);
    __builtin_amdgcn_global_load_async_to_lds_b128(to_glb(g), to_lds(l), 48, 0);
    wait_async0();
#else
#pragma unroll
    for (int j = 0; j < 4; ++j)
      *(v8h*)(l + 8 * j) = *(const v8h*)(g + 8 * j);
#endif
  }
  __syncthreads();
  int c = tid >> 2, pb = (tid & 3) * 32;
  _Float16* dst = vt + ((tokbase >> 1) + (long)c * 128) + pb;  // (b*128+y)*64 + c*128
#pragma unroll
  for (int j = 0; j < 4; ++j) {
    v8h h;
#pragma unroll
    for (int ee = 0; ee < 8; ++ee) h[ee] = sT[(pb + j * 8 + ee) * 72 + c];
    *(v8h*)(dst + 8 * j) = h;
  }
}

// ---------------------------------------------------------------------------
// Attention: block per (b, head=y); seq=128, c=64.
// S = Q Kt straight from global qkv (all-wide frags); softmax in registers;
// P -> LDS; O computed transposed (A = Vt) so PV frags and O stores are wide.
// ---------------------------------------------------------------------------
__global__ __launch_bounds__(256) void attn_kernel(
    const _Float16* __restrict__ qkv, const _Float16* __restrict__ vt,
    _Float16* __restrict__ obuf, float scale) {
  __shared__ __align__(16) _Float16 sP[128 * 136];
  const int bb = blockIdx.x >> 7;
  const int y = blockIdx.x & 127;
  const long tokbase = ((long)bb * 128 + y) * 128;
  const long vtbase = tokbase >> 1;  // (b*128+y)*64

  const int tid = threadIdx.x;
  const int lane = tid & 31, wave = tid >> 5;
  const int ln = lane & 15, hf = lane >> 4;
  const int qm = wave * 16;

  v16h aq[2];
#pragma unroll
  for (int kc = 0; kc < 2; ++kc)
    aq[kc] = frag_a(qkv + (tokbase + qm + ln) * CDIM + kc * 32, hf);

  v8f s[8];
#pragma unroll
  for (int nt = 0; nt < 8; ++nt) {
    v8f acc = {};
#pragma unroll
    for (int kc = 0; kc < 2; ++kc) {
      v16h b = frag_b(qkv + (tokbase + nt * 16 + ln) * CDIM + HD + kc * 32, hf);
      acc = wmma(aq[kc], b, acc);
    }
    s[nt] = acc;
  }

#pragma unroll
  for (int r = 0; r < 8; ++r) {
    float mx = -3.0e38f;
#pragma unroll
    for (int nt = 0; nt < 8; ++nt) mx = fmaxf(mx, s[nt][r]);
#pragma unroll
    for (int xm = 8; xm >= 1; xm >>= 1) mx = fmaxf(mx, __shfl_xor(mx, xm, 32));
    float sum = 0.f;
#pragma unroll
    for (int nt = 0; nt < 8; ++nt) {
      float p = __expf((s[nt][r] - mx) * scale);
      s[nt][r] = p;
      sum += p;
    }
#pragma unroll
    for (int xm = 8; xm >= 1; xm >>= 1) sum += __shfl_xor(sum, xm, 32);
    float inv = 1.0f / sum;
#pragma unroll
    for (int nt = 0; nt < 8; ++nt) s[nt][r] *= inv;
  }

#pragma unroll
  for (int nt = 0; nt < 8; ++nt)
#pragma unroll
    for (int r = 0; r < 8; ++r)
      sP[(qm + r + (hf << 3)) * 136 + nt * 16 + ln] = (_Float16)(s[nt][r]);
  __syncthreads();

  // O^T = Vt x Pt : M=c (4 tiles), this wave's N tile = its own 16 q-rows
  v8f o[4] = {};
#pragma unroll
  for (int kc = 0; kc < 4; ++kc) {
    v16h b = frag_b(sP + (wave * 16 + ln) * 136 + kc * 32, hf);
#pragma unroll
    for (int tm = 0; tm < 4; ++tm) {
      v16h a = frag_a(vt + (vtbase + tm * 16 + ln) * 128 + kc * 32, hf);
      o[tm] = wmma(a, b, o[tm]);
    }
  }
  int qpos = wave * 16 + ln;
  long tokp = ((long)bb * 128 + qpos) * 128 + y;  // swapaxes order
#pragma unroll
  for (int tm = 0; tm < 4; ++tm) {
    v8h h;
#pragma unroll
    for (int r = 0; r < 8; ++r) h[r] = (_Float16)o[tm][r];
    *(v8h*)(obuf + tokp * HD + tm * 16 + 8 * hf) = h;
  }
}

// ---------------------------------------------------------------------------
// Host orchestration
// ---------------------------------------------------------------------------
extern "C" void kernel_launch(void* const* d_in, const int* in_sizes, int n_in,
                              void* d_out, int out_size, void* d_ws,
                              size_t ws_size, hipStream_t stream) {
  const float* x        = (const float*)d_in[0];
  const float* proj1_w  = (const float*)d_in[1];
  const float* proj1_b  = (const float*)d_in[2];
  const float* conv_w[6] = {(const float*)d_in[3],  (const float*)d_in[5],
                            (const float*)d_in[7],  (const float*)d_in[9],
                            (const float*)d_in[11], (const float*)d_in[13]};
  const float* conv_b[6] = {(const float*)d_in[4],  (const float*)d_in[6],
                            (const float*)d_in[8],  (const float*)d_in[10],
                            (const float*)d_in[12], (const float*)d_in[14]};
  const float* lin_w    = (const float*)d_in[15];
  const float* lin_b    = (const float*)d_in[16];
  const float* router_w = (const float*)d_in[17];
  const float* router_b = (const float*)d_in[18];
  const float* moe_w1   = (const float*)d_in[19];
  const float* moe_b1   = (const float*)d_in[20];
  const float* moe_w2   = (const float*)d_in[21];
  const float* moe_b2   = (const float*)d_in[22];
  const float* qkv_w    = (const float*)d_in[23];
  const float* ap_w     = (const float*)d_in[24];
  const float* ap_b     = (const float*)d_in[25];
  const float* proj3_w  = (const float*)d_in[26];
  const float* proj3_b  = (const float*)d_in[27];

  char* ws = (char*)d_ws;
  size_t off = 0;
  auto alloc = [&](size_t bytes) {
    void* p = ws + off;
    off += (bytes + 255) & ~(size_t)255;
    return p;
  };
  _Float16* x16      = (_Float16*)alloc((size_t)NTOK * CDIM * 2);
  _Float16* proj1buf = (_Float16*)alloc((size_t)NTOK * CDIM * 2);
  _Float16* catbuf   = (_Float16*)alloc((size_t)NTOK * 128 * 2);
  _Float16* tbuf     = (_Float16*)alloc((size_t)NTOK * HD * 2);
  _Float16* t2buf    = (_Float16*)alloc((size_t)NTOK * HD * 2);
  _Float16* qkvbuf   = (_Float16*)alloc((size_t)NTOK * CDIM * 2);
  _Float16* vtbuf    = (_Float16*)alloc((size_t)NTOK * HD * 2);
  _Float16* obuf     = (_Float16*)alloc((size_t)NTOK * HD * 2);
  _Float16* xcat     = (_Float16*)alloc((size_t)NTOK * CDIM * 2);
  float*    gatebuf  = (float*)alloc((size_t)NTOK * 4);
  int*      listbuf  = (int*)alloc((size_t)NEXP * NTOK * 4);
  int*      cntbuf   = (int*)alloc(128);
  _Float16* proj1_wt = (_Float16*)alloc(192 * 192 * 2);
  _Float16* proj3_wt = (_Float16*)alloc(192 * 192 * 2);
  _Float16* lin_wt   = (_Float16*)alloc(3 * 64 * 128 * 2);
  _Float16* qkv_wt   = (_Float16*)alloc(3 * 192 * 64 * 2);
  _Float16* ap_wt    = (_Float16*)alloc(3 * 64 * 64 * 2);
  _Float16* conv_wt  = (_Float16*)alloc(6 * 64 * 576 * 2);
  _Float16* w1t      = (_Float16*)alloc(15 * 256 * 64 * 2);
  _Float16* w2t      = (_Float16*)alloc(15 * 64 * 256 * 2);

  const int mtiles = NTOK / 64;                // 2048
  const float scale = 0.35355339059327373f;    // 1/sqrt(8)

  auto swz = [&](const float* W, _Float16* Wt, int K, int N) {
    wswz_kernel<<<(K * N + 255) / 256, 256, 0, stream>>>(W, Wt, K, N);
  };

  zero_counts_kernel<<<1, 32, 0, stream>>>(cntbuf, 3 * 8);
  cvt16_kernel<<<NTOK * CDIM / 2048, 256, 0, stream>>>(x, x16);

  swz(proj1_w, proj1_wt, CDIM, CDIM);
  swz(proj3_w, proj3_wt, CDIM, CDIM);
  for (int i = 0; i < 3; ++i) {
    swz(lin_w + (size_t)i * 128 * HD, lin_wt + (size_t)i * HD * 128, 128, HD);
    swz(qkv_w + (size_t)i * HD * CDIM, qkv_wt + (size_t)i * CDIM * HD, HD, CDIM);
    swz(ap_w + (size_t)i * HD * HD, ap_wt + (size_t)i * HD * HD, HD, HD);
    for (int e = 0; e < NEXP; ++e) {
      size_t ofs = (size_t)(i * NEXP + e) * HD * HIDDEN;
      swz(moe_w1 + ofs, w1t + ofs, HD, HIDDEN);
      swz(moe_w2 + ofs, w2t + ofs, HIDDEN, HD);
    }
  }
  for (int ci = 0; ci < 6; ++ci)
    convswz_kernel<<<144, 256, 0, stream>>>(conv_w[ci],
                                            conv_wt + (size_t)ci * 64 * 576);

  // proj1
  gemm_tn_kernel<<<dim3(mtiles, 3), 256, 0, stream>>>(
      x16, CDIM, proj1_wt, proj1_b, proj1buf, CDIM, 0, CDIM, 1);

  const int cs[6][6] = {{3, 3, 1, 1, 1, 1}, {3, 3, 2, 2, 2, 2},
                        {1, 9, 1, 1, 0, 4}, {1, 9, 1, 2, 0, 8},
                        {9, 1, 1, 1, 4, 0}, {9, 1, 2, 1, 8, 0}};

  for (int i = 0; i < 3; ++i) {
    for (int j = 0; j < 2; ++j) {
      int ci = i * 2 + j;
      conv_tn_kernel<<<2048, 256, 0, stream>>>(
          proj1buf, i * HD, conv_wt + (size_t)ci * 64 * 576, conv_b[ci],
          catbuf, j * HD, cs[ci][0], cs[ci][1], cs[ci][2], cs[ci][3], cs[ci][4],
          cs[ci][5]);
    }
    gemm_tn_kernel<<<dim3(mtiles, 1), 256, 0, stream>>>(
        catbuf, 128, lin_wt + (size_t)i * HD * 128, lin_b + i * HD, tbuf, HD, 0,
        128, 1);
    router_kernel<<<NTOK / 256, 256, 0, stream>>>(
        tbuf, router_w + (size_t)i * HD * NEXP, router_b + i * NEXP, gatebuf,
        listbuf, cntbuf + i * 8, NTOK);
    moe_kernel<<<NEXP * mtiles, 256, 0, stream>>>(
        tbuf, listbuf, cntbuf + i * 8, gatebuf,
        w1t + (size_t)i * NEXP * HD * HIDDEN, moe_b1 + i * NEXP * HIDDEN,
        w2t + (size_t)i * NEXP * HIDDEN * HD, moe_b2 + i * NEXP * HD, t2buf,
        NTOK, mtiles);
    gemm_tn_kernel<<<dim3(mtiles, 3), 256, 0, stream>>>(
        t2buf, HD, qkv_wt + (size_t)i * CDIM * HD, (const float*)nullptr,
        qkvbuf, CDIM, 0, HD, 1);
    vtrans_kernel<<<1024, 256, 0, stream>>>(qkvbuf, vtbuf);
    attn_kernel<<<1024, 256, 0, stream>>>(qkvbuf, vtbuf, obuf, scale);
    gemm_tn_kernel<<<dim3(mtiles, 1), 256, 0, stream>>>(
        obuf, HD, ap_wt + (size_t)i * HD * HD, ap_b + i * HD, xcat, CDIM,
        i * HD, HD, 1);
  }

  gemm_tn_kernel<<<dim3(mtiles, 3), 256, 0, stream>>>(
      xcat, CDIM, proj3_wt, proj3_b, d_out, CDIM, 0, CDIM, 0);
}
